// TensorBoard_29729763622957
// MI455X (gfx1250) — matrix operand order repaired and checked by reading, not compiled, not run on hardware
//
#include <hip/hip_runtime.h>
#include <hip/hip_bf16.h>

// ---------------------------------------------------------------------------
// Go board batch: group labels + liberties + legal + scores + WMMA segment pool
// B=2048 boards, 19x19=361 cells, D=64 features.
// Output (concat, float): out_feats[B*361*64], libs[B*361], legal[B*361], scores[B*2]
// ---------------------------------------------------------------------------

static constexpr int kB  = 2048;
static constexpr int kW  = 19;
static constexpr int kN  = 361;   // 19*19
static constexpr int kD  = 64;
static constexpr int kNP = 368;   // rows padded to 16
static constexpr int kKP = 384;   // K padded to 32
static constexpr int kKPitch = 392; // transposed-LDS K pitch (bank-conflict dodge)

typedef __attribute__((ext_vector_type(16))) __bf16 v16bf;
typedef __attribute__((ext_vector_type(8)))  __bf16 v8bf;
typedef __attribute__((ext_vector_type(8)))  float  v8f;

// ===========================================================================
// Kernel 1: per-board labeling, liberties, legal, scores; emits keys to ws.
// One block per board, 384 threads.
// ===========================================================================
__global__ __launch_bounds__(384) void go_label_kernel(
    const unsigned char* __restrict__ stones,   // [B,2,361] bool
    const int* __restrict__ ko,                 // [B,2]
    float* __restrict__ out_libs,               // [B,361]
    float* __restrict__ out_legal,              // [B,361]
    float* __restrict__ out_scores,             // [B,2]
    int* __restrict__ keys_ws)                  // [B,384]
{
    const int b   = blockIdx.x;
    const int tid = threadIdx.x;

    __shared__ signed char   col[kN];
    __shared__ int           lab[kN];
    __shared__ int           tmp[kN];
    __shared__ int           libs[kN];
    __shared__ unsigned char legal[kN];
    __shared__ int           sc[2];
    __shared__ int           s_changed;

    const unsigned char* sb = stones + (size_t)b * 2 * kN;

    if (tid < kN) {
        const bool black = sb[tid] != 0;
        const bool white = (sb[kN + tid] != 0) && !black;
        col[tid]   = black ? 0 : (white ? 1 : -1);
        lab[tid]   = tid;
        libs[tid]  = 0;
        legal[tid] = (!black && !white) ? 1 : 0;
    }
    if (tid == 0) { sc[0] = 0; sc[1] = 0; }
    __syncthreads();

    if (tid < kN) {
        if (col[tid] == 0)      atomicAdd(&sc[0], 1);
        else if (col[tid] == 1) atomicAdd(&sc[1], 1);
    }

    // --- iterative min-propagation + one pointer-jump per round (matches ref)
    for (;;) {
        if (tid == 0) s_changed = 0;
        __syncthreads();

        int m = 0;
        if (tid < kN) {
            m = lab[tid];
            const int r = tid / kW, c = tid % kW;
            const int myc = col[tid];
            if (r > 0      && col[tid - kW] == myc) m = min(m, lab[tid - kW]);
            if (r < kW - 1 && col[tid + kW] == myc) m = min(m, lab[tid + kW]);
            if (c > 0      && col[tid - 1 ] == myc) m = min(m, lab[tid - 1 ]);
            if (c < kW - 1 && col[tid + 1 ] == myc) m = min(m, lab[tid + 1 ]);
        }
        __syncthreads();
        if (tid < kN) tmp[tid] = m;
        __syncthreads();
        bool ch = false;
        if (tid < kN) {
            const int nm = min(m, tmp[m]);   // pointer jump (m always < kN)
            if (nm != lab[tid]) { lab[tid] = nm; ch = true; }
        }
        if (ch) s_changed = 1;
        __syncthreads();
        if (!s_changed) break;
    }

    // --- liberties: empty cells credit each distinct adjacent stone-group once
    if (tid < kN && col[tid] == -1) {
        const int r = tid / kW, c = tid % kW;
        int li[4]; // reference dir order: up, down, left, right
        li[0] = (r > 0      && col[tid - kW] >= 0) ? lab[tid - kW] : kN;
        li[1] = (r < kW - 1 && col[tid + kW] >= 0) ? lab[tid + kW] : kN;
        li[2] = (c > 0      && col[tid - 1 ] >= 0) ? lab[tid - 1 ] : kN;
        li[3] = (c < kW - 1 && col[tid + 1 ] >= 0) ? lab[tid + 1 ] : kN;
        #pragma unroll
        for (int i = 0; i < 4; ++i) {
            bool dup = false;
            #pragma unroll
            for (int j = 0; j < 4; ++j)
                if (j < i) dup = dup || (li[i] == li[j]);
            if (li[i] < kN && !dup) atomicAdd(&libs[li[i]], 1);
        }
    }
    __syncthreads();

    // --- ko point
    if (tid == 0) {
        const int kr = ko[b * 2 + 0];
        const int kc = ko[b * 2 + 1];
        if (kr >= 0) {
            const int rr = min(max(kr, 0), kW - 1);
            const int cc = min(max(kc, 0), kW - 1);
            legal[rr * kW + cc] = 0;
        }
    }
    __syncthreads();

    // --- write outputs + keys for the WMMA pool kernel
    if (tid < kN) {
        out_libs [(size_t)b * kN + tid] = (float)libs[tid];
        out_legal[(size_t)b * kN + tid] = (float)legal[tid];
        // key: group label for stones (0..360), unique non-matchable id for empties
        keys_ws[b * kKP + tid] = (col[tid] >= 0) ? lab[tid] : (kN + tid);
    } else if (tid < kKP) {
        keys_ws[b * kKP + tid] = -5;   // padding rows: never match
    }
    if (tid < 2) out_scores[(size_t)b * 2 + tid] = (float)sc[tid];
}

// ===========================================================================
// Kernel 2: out = M * F per board via bf16 WMMA, M[p,q]=stone(p)&&key(p)==key(q)
// A synthesized per lane from key compares; B read as contiguous ds_load_b128
// from a transposed LDS image of F. One block per board, 256 threads (8 waves).
// Padded GEMM: 368 x 384 x 64. Per K-step: all 4 B-fragment load pairs issued
// first, A built while they are in flight, then 4 back-to-back WMMAs.
// ===========================================================================
__global__ __launch_bounds__(256) void go_pool_wmma_kernel(
    const float* __restrict__ features,   // [B,361,64] f32
    const int*   __restrict__ keys_ws,    // [B,384]
    float*       __restrict__ out_feats)  // [B,361,64] f32
{
    const int b   = blockIdx.x;
    const int tid = threadIdx.x;

    // transposed feature image: fT[col][k], contiguous in k per column
    __shared__ __align__(16) __bf16 fT[kD * kKPitch];
    __shared__ __align__(16) int    kLDS[kKP];

    // stage features -> transposed LDS (bf16) with float4 global loads
    for (int i = tid; i < (kKP * kD) / 4; i += 256) {
        const int flat = i * 4;
        const int r = flat >> 6;       // k row
        const int c = flat & 63;       // feature col (multiple of 4)
        float4 v = make_float4(0.f, 0.f, 0.f, 0.f);
        if (r < kN)
            v = ((const float4*)(features + ((size_t)b * kN + r) * kD))[c >> 2];
        fT[(c + 0) * kKPitch + r] = (__bf16)v.x;
        fT[(c + 1) * kKPitch + r] = (__bf16)v.y;
        fT[(c + 2) * kKPitch + r] = (__bf16)v.z;
        fT[(c + 3) * kKPitch + r] = (__bf16)v.w;
    }
    for (int i = tid; i < kKP; i += 256) kLDS[i] = keys_ws[b * kKP + i];
    __syncthreads();

    const int w    = tid >> 5;        // wave id (wave32)
    const int lane = tid & 31;
    const int h    = lane >> 4;       // lane half
    const int ln   = lane & 15;

    const __bf16 one  = (__bf16)1.0f;
    const __bf16 zero = (__bf16)0.0f;

    float* const outb = out_feats + (size_t)b * kN * kD;

    // 23 row tiles over 8 waves; each row tile carries all 4 column tiles
    for (int rt = w; rt < kNP / 16; rt += 8) {
        const int  mrow   = rt * 16 + ln;                  // A row this lane owns
        const int  keyP   = (mrow < kN) ? kLDS[mrow] : -7;
        const bool stoneP = (keyP >= 0) && (keyP < kN);    // stone iff key is a label

        v8f acc[4] = {v8f{}, v8f{}, v8f{}, v8f{}};

        for (int kt = 0; kt < kKP / 32; ++kt) {
            const int kb = kt * 32;

            // ---- issue ALL B-fragment LDS loads first (8x ds_load_b128) ----
            v8bf blo[4], bhi[4];
            #pragma unroll
            for (int ct = 0; ct < 4; ++ct) {
                const __bf16* bp = &fT[(ct * 16 + ln) * kKPitch + kb + h * 8];
                blo[ct] = *(const v8bf*)(bp);      // ds_load_b128
                bhi[ct] = *(const v8bf*)(bp + 16); // ds_load_b128
            }

            // ---- build A while the DS loads are in flight ----
            // lanes 0-15: K {0..7,16..23}; lanes 16-31: K {8..15,24..31}
            const int4 q0 = ((const int4*)&kLDS[kb +      h * 8])[0];
            const int4 q1 = ((const int4*)&kLDS[kb +      h * 8])[1];
            const int4 q2 = ((const int4*)&kLDS[kb + 16 + h * 8])[0];
            const int4 q3 = ((const int4*)&kLDS[kb + 16 + h * 8])[1];

            v16bf a;
            a[0]  = (stoneP && keyP == q0.x) ? one : zero;
            a[1]  = (stoneP && keyP == q0.y) ? one : zero;
            a[2]  = (stoneP && keyP == q0.z) ? one : zero;
            a[3]  = (stoneP && keyP == q0.w) ? one : zero;
            a[4]  = (stoneP && keyP == q1.x) ? one : zero;
            a[5]  = (stoneP && keyP == q1.y) ? one : zero;
            a[6]  = (stoneP && keyP == q1.z) ? one : zero;
            a[7]  = (stoneP && keyP == q1.w) ? one : zero;
            a[8]  = (stoneP && keyP == q2.x) ? one : zero;
            a[9]  = (stoneP && keyP == q2.y) ? one : zero;
            a[10] = (stoneP && keyP == q2.z) ? one : zero;
            a[11] = (stoneP && keyP == q2.w) ? one : zero;
            a[12] = (stoneP && keyP == q3.x) ? one : zero;
            a[13] = (stoneP && keyP == q3.y) ? one : zero;
            a[14] = (stoneP && keyP == q3.z) ? one : zero;
            a[15] = (stoneP && keyP == q3.w) ? one : zero;

            // ---- 4 back-to-back WMMAs reusing the A fragment ----
            #pragma unroll
            for (int ct = 0; ct < 4; ++ct) {
                const v16bf bf = __builtin_shufflevector(
                    blo[ct], bhi[ct],
                    0, 1, 2, 3, 4, 5, 6, 7, 8, 9, 10, 11, 12, 13, 14, 15);
                acc[ct] = __builtin_amdgcn_wmma_f32_16x16x32_bf16(
                    /*neg_a=*/false, a, /*neg_b=*/false, bf,
                    /*c_mod=*/(short)0, acc[ct], /*reuse_a=*/false, /*reuse_b=*/false);
            }
        }

        // D layout: VGPR r holds M = rt*16 + r + 8*h, N = ct*16 + ln
        if (rt < 22) {
            // full tile: unconditional streaming stores (write-once, skip L2 reuse)
            #pragma unroll
            for (int ct = 0; ct < 4; ++ct) {
                float* const p0 = outb + (size_t)(rt * 16 + 8 * h) * kD + ct * 16 + ln;
                #pragma unroll
                for (int r = 0; r < 8; ++r)
                    __builtin_nontemporal_store(acc[ct][r], p0 + (size_t)r * kD);
            }
        } else {
            #pragma unroll
            for (int ct = 0; ct < 4; ++ct) {
                const int ncol = ct * 16 + ln;
                #pragma unroll
                for (int r = 0; r < 8; ++r) {
                    const int m = rt * 16 + r + 8 * h;
                    if (m < kN)
                        __builtin_nontemporal_store(acc[ct][r],
                                                    outb + (size_t)m * kD + ncol);
                }
            }
        }
    }
}

// ===========================================================================
extern "C" void kernel_launch(void* const* d_in, const int* in_sizes, int n_in,
                              void* d_out, int out_size, void* d_ws, size_t ws_size,
                              hipStream_t stream) {
    const unsigned char* stones   = (const unsigned char*)d_in[0]; // bool [B,2,19,19]
    const int*           ko       = (const int*)d_in[1];           // [B,2]
    // d_in[2] current_player: unused by the reference outputs
    const float*         features = (const float*)d_in[3];         // [B,361,64]

    float* out        = (float*)d_out;
    float* out_feats  = out;                                        // B*361*64
    float* out_libs   = out_feats + (size_t)kB * kN * kD;           // B*361
    float* out_legal  = out_libs  + (size_t)kB * kN;                // B*361
    float* out_scores = out_legal + (size_t)kB * kN;                // B*2

    int* keys_ws = (int*)d_ws;                                      // B*384 ints

    go_label_kernel<<<kB, 384, 0, stream>>>(stones, ko, out_libs, out_legal,
                                            out_scores, keys_ws);
    go_pool_wmma_kernel<<<kB, 256, 0, stream>>>(features, keys_ws, out_feats);
}
